// ScoreModel_27384711480159
// MI455X (gfx1250) — matrix operand order repaired and checked by reading, not compiled, not run on hardware
//
#include <hip/hip_runtime.h>
#include <hip/hip_bf16.h>
#include <math.h>

typedef __attribute__((ext_vector_type(16))) _Float16 v16h;
typedef __attribute__((ext_vector_type(8)))  _Float16 v8h;
typedef __attribute__((ext_vector_type(8)))  float    v8f;

#define SQ3F 1.7320508075688772f
#define SQ2F 1.4142135623730951f

// Load a WMMA A-operand fragment from f16 LDS: per lane, two contiguous
// 16-byte runs (elements 0..7 = K base+0..7, elements 8..15 = K base+16..23).
__device__ inline v16h frag_ld(const _Float16* p) {
  v8h lo = *(const v8h*)(p);
  v8h hi = *(const v8h*)(p + 16);
  v16h r;
#pragma unroll
  for (int q = 0; q < 8; ++q) { r[q] = lo[q]; r[q + 8] = hi[q]; }
  return r;
}

// ---------------------------------------------------------------------------
// timestep embedding: sig[n][0..15]=sin(t*f_j), sig[n][16..31]=cos(t*f_j)
// ---------------------------------------------------------------------------
__global__ void k_sig(const float* __restrict__ t, float* __restrict__ sig, int N) {
  int n = blockIdx.x * blockDim.x + threadIdx.x;
  if (n >= N) return;
  float tv = t[n];
  const float c = -logf(10000.0f) / 16.0f;
#pragma unroll
  for (int j = 0; j < 16; ++j) {
    float f = expf(c * (float)j);
    float ang = tv * f;
    sig[n * 32 + j]      = sinf(ang);
    sig[n * 32 + 16 + j] = cosf(ang);
  }
}

// ---------------------------------------------------------------------------
// node MLP: x[n][:16] = relu([x_feat(64), sig(32)] @ W1(96x16) + b1) @ W2 + b2
// pads x[n][16..111] with zero.
// ---------------------------------------------------------------------------
__global__ void k_node_mlp(const float* __restrict__ xf, const float* __restrict__ sig,
                           const float* __restrict__ W1, const float* __restrict__ b1,
                           const float* __restrict__ W2, const float* __restrict__ b2,
                           float* __restrict__ x, int N) {
  int n = blockIdx.x * blockDim.x + threadIdx.x;
  if (n >= N) return;
  float h[16];
#pragma unroll
  for (int j = 0; j < 16; ++j) h[j] = b1[j];
  for (int k = 0; k < 96; ++k) {
    float v = (k < 64) ? xf[(size_t)n * 64 + k] : sig[(size_t)n * 32 + (k - 64)];
#pragma unroll
    for (int j = 0; j < 16; ++j) h[j] += v * W1[k * 16 + j];
  }
#pragma unroll
  for (int j = 0; j < 16; ++j) h[j] = fmaxf(h[j], 0.0f);
  for (int j = 0; j < 16; ++j) {
    float o = b2[j];
#pragma unroll
    for (int k = 0; k < 16; ++k) o += h[k] * W2[k * 16 + j];
    x[(size_t)n * 112 + j] = o;
  }
  for (int j = 16; j < 112; ++j) x[(size_t)n * 112 + j] = 0.0f;
}

// ---------------------------------------------------------------------------
// edge attributes: gaussian smearing + sig concat -> MLP 64->16->16, plus sph_l1
// sig_first: [sig, gs] vs [gs, sig].   sig_src: take sig of source node.
// ---------------------------------------------------------------------------
__global__ void k_edge_attr(const float* __restrict__ pos, const float* __restrict__ dpos,
                            const float* __restrict__ sig,
                            const int* __restrict__ srci, const int* __restrict__ dsti,
                            const float* __restrict__ W1, const float* __restrict__ b1,
                            const float* __restrict__ W2, const float* __restrict__ b2,
                            float* __restrict__ attr, float* __restrict__ sh,
                            int E, int sig_first, int sig_src) {
  int e = blockIdx.x * blockDim.x + threadIdx.x;
  if (e >= E) return;
  int s = srci[e], d = dsti[e];
  float px = pos[s * 3 + 0] - dpos[d * 3 + 0];
  float py = pos[s * 3 + 1] - dpos[d * 3 + 1];
  float pz = pos[s * 3 + 2] - dpos[d * 3 + 2];
  float dist = sqrtf(px * px + py * py + pz * pz);
  float nn = fmaxf(dist, 1e-12f);
  sh[(size_t)e * 4 + 0] = 1.0f;
  sh[(size_t)e * 4 + 1] = SQ3F * px / nn;
  sh[(size_t)e * 4 + 2] = SQ3F * py / nn;
  sh[(size_t)e * 4 + 3] = SQ3F * pz / nn;
  const float step = 10.0f / 31.0f;
  const float coeff = -0.5f / (step * step);
  int sn = sig_src ? s : d;
  float h[16];
#pragma unroll
  for (int j = 0; j < 16; ++j) h[j] = b1[j];
  for (int k = 0; k < 64; ++k) {
    float v;
    if (sig_first) {
      if (k < 32) v = sig[(size_t)sn * 32 + k];
      else { float dd = dist - (float)(k - 32) * step; v = expf(coeff * dd * dd); }
    } else {
      if (k < 32) { float dd = dist - (float)k * step; v = expf(coeff * dd * dd); }
      else v = sig[(size_t)sn * 32 + (k - 32)];
    }
#pragma unroll
    for (int j = 0; j < 16; ++j) h[j] += v * W1[k * 16 + j];
  }
#pragma unroll
  for (int j = 0; j < 16; ++j) h[j] = fmaxf(h[j], 0.0f);
  for (int j = 0; j < 16; ++j) {
    float o = b2[j];
#pragma unroll
    for (int k = 0; k < 16; ++k) o += h[k] * W2[k * 16 + j];
    attr[(size_t)e * 16 + j] = o;
  }
}

// ---------------------------------------------------------------------------
// pack weight matrix W (K x ncols, f32, row-major) into f16 WMMA B-fragments:
// out[((t*ksteps + ks)*32 + lane)*16 + e] = W[ks*32 + (lane/16)*8 + (e%8) + (e/8)*16][t*16 + lane%16]
// ---------------------------------------------------------------------------
__global__ void k_pack(const float* __restrict__ W, int K, int ncols, int ksteps,
                       _Float16* __restrict__ out, int total) {
  int id = blockIdx.x * blockDim.x + threadIdx.x;
  if (id >= total) return;
  int e = id & 15;
  int lane = (id >> 4) & 31;
  int ks = (id >> 9) % ksteps;
  int t = id / (512 * ksteps);
  int col = t * 16 + (lane & 15);
  int k = ks * 32 + (lane >> 4) * 8 + (e & 7) + (e >> 3) * 16;
  float v = (k < K) ? W[(size_t)k * ncols + col] : 0.0f;
  out[id] = (_Float16)v;
}

// ---------------------------------------------------------------------------
// fused per-edge conv: w = relu(ea@Wa+ba)@Wb+bb (WMMA), then tensor product
// tp0 (L=0) / tp1 (L=1), scatter-add into acc[dst] + counts.
// One wave = 16 edges. N-tile t of w == (b = t/16, u = t%16) of the per-edge
// tensor-product matrices, so TP accumulators live in fixed lane registers.
// A/H operands staged in f16 LDS rows of 72 halves (144B, 16B-aligned, K
// padded 48->64 with zeros) so each fragment is two ds_load_b128, no cvt.
// ---------------------------------------------------------------------------
template <int L>
__global__ __launch_bounds__(128) void k_conv(
    const float* __restrict__ x, const float* __restrict__ eattr, const float* __restrict__ esh,
    const int* __restrict__ srci, const int* __restrict__ dsti,
    const _Float16* __restrict__ Wa, const float* __restrict__ ba,
    const _Float16* __restrict__ Wb, const float* __restrict__ bb,
    float* __restrict__ acc, float* __restrict__ cnt, int E) {
  constexpr int NUMEL = (L == 0) ? 512 : 1280;
  constexpr int NT = NUMEL / 16;
  // per-wave floats: EAh(576) Hh(576) S(256) SH(64) [VO(768) if L==1]
  constexpr int WF = 576 + 576 + 256 + 64 + ((L == 1) ? 768 : 0);
  __shared__ __align__(16) float lds[4 * WF];
  const int lane = threadIdx.x & 31;
  const int wv = threadIdx.x >> 5;
  _Float16* EAh = (_Float16*)(lds + wv * WF);          // [16][72] halves
  _Float16* Hh  = (_Float16*)(lds + wv * WF + 576);    // [16][72] halves
  float* S   = lds + wv * WF + 1152;                   // [16][16] s = x[src][:16]
  float* SHm = lds + wv * WF + 1408;                   // [16][4]
  float* VO  = lds + wv * WF + 1472;                   // [16][16][3] (L==1)
  int ntiles = (E + 15) >> 4;
  int tile = blockIdx.x * 4 + wv;
  bool active = (tile < ntiles);
  if (!active) tile = ntiles - 1;
  const int e0 = tile * 16;

  for (int idx = lane; idx < 768; idx += 32) {
    int m = idx / 48, k = idx - m * 48;
    int e = e0 + m; if (e >= E) e = E - 1;
    float v;
    if (k < 16)      v = eattr[(size_t)e * 16 + k];
    else if (k < 32) { v = x[(size_t)srci[e] * 112 + (k - 16)]; S[m * 16 + (k - 16)] = v; }
    else             v = x[(size_t)dsti[e] * 112 + (k - 32)];
    EAh[m * 72 + k] = (_Float16)v;
  }
  for (int idx = lane; idx < 256; idx += 32) {  // zero K-pad cols 48..63
    int m = idx >> 4, k = 48 + (idx & 15);
    EAh[m * 72 + k] = (_Float16)0.0f;
    Hh[m * 72 + k]  = (_Float16)0.0f;
  }
  for (int idx = lane; idx < 64; idx += 32) {
    int m = idx >> 2, c = idx & 3;
    int e = e0 + m; if (e >= E) e = E - 1;
    SHm[idx] = esh[(size_t)e * 4 + c];
  }
  if (L == 1) {
    for (int idx = lane; idx < 768; idx += 32) {
      int m = idx / 48, k = idx - m * 48;
      int e = e0 + m; if (e >= E) e = E - 1;
      VO[idx] = x[(size_t)srci[e] * 112 + 16 + k];
    }
  }
  if (lane < 16) {
    int e = e0 + lane;
    if (active && e < E) atomicAdd(&cnt[dsti[e]], 1.0f);
  }
  __syncthreads();

  const int mrow = lane & 15;
  const int khalf = (lane >> 4) * 8;
  const int mbase = (lane < 16) ? 0 : 8;
  const int ncol = lane & 15;

  // A fragments of ea (K=48 padded to 64 -> 2 k-steps)
  v16h a0 = frag_ld(EAh + mrow * 72 + khalf);
  v16h a1 = frag_ld(EAh + mrow * 72 + 32 + khalf);
  // stage 1: h = relu(ea @ Wa + ba), 48 cols = 3 N-tiles
  for (int nt = 0; nt < 3; ++nt) {
    v8f c = {0.f, 0.f, 0.f, 0.f, 0.f, 0.f, 0.f, 0.f};
    const v16h b0 = *(const v16h*)(Wa + ((size_t)(nt * 2 + 0) * 32 + lane) * 16);
    const v16h b1 = *(const v16h*)(Wa + ((size_t)(nt * 2 + 1) * 32 + lane) * 16);
    c = __builtin_amdgcn_wmma_f32_16x16x32_f16(false, a0, false, b0, (short)0, c, false, false);
    c = __builtin_amdgcn_wmma_f32_16x16x32_f16(false, a1, false, b1, (short)0, c, false, false);
    float bias = ba[nt * 16 + ncol];
#pragma unroll
    for (int j = 0; j < 8; ++j)
      Hh[(mbase + j) * 72 + nt * 16 + ncol] = (_Float16)fmaxf(c[j] + bias, 0.0f);
  }
  __syncthreads();
  v16h h0 = frag_ld(Hh + mrow * 72 + khalf);
  v16h h1 = frag_ld(Hh + mrow * 72 + 32 + khalf);

  float acc_s[8];
  float acc_v1[8][3];
  float acc_v2[8][3];
#pragma unroll
  for (int j = 0; j < 8; ++j) {
    acc_s[j] = 0.f;
#pragma unroll
    for (int i = 0; i < 3; ++i) { acc_v1[j][i] = 0.f; acc_v2[j][i] = 0.f; }
  }
  const float A0 = (L == 0) ? 0.25f : 0.17677669529663687f;  // 1/sqrt16, 1/sqrt32
  const float A2 = 0.25f;                                    // 1/sqrt16

  for (int t = 0; t < NT; ++t) {
    v8f c = {0.f, 0.f, 0.f, 0.f, 0.f, 0.f, 0.f, 0.f};
    const _Float16* bp = Wb + ((size_t)(t * 2) * 32 + lane) * 16;
    __builtin_prefetch(bp + 2048, 0, 1);  // next-next tile (speculative)
    const v16h b0 = *(const v16h*)bp;
    const v16h b1 = *(const v16h*)(bp + 512);
    c = __builtin_amdgcn_wmma_f32_16x16x32_f16(false, h0, false, b0, (short)0, c, false, false);
    c = __builtin_amdgcn_wmma_f32_16x16x32_f16(false, h1, false, b1, (short)0, c, false, false);
    float bias = bb[t * 16 + ncol];
    int u = t & 15;    // matrix row (uniform per tile)
    int b = t >> 4;    // which per-edge matrix (uniform per tile)
#pragma unroll
    for (int j = 0; j < 8; ++j) {
      int m = mbase + j;
      float w = c[j] + bias;
      float su = S[m * 16 + u];
      float ss = SHm[m * 4 + 0];
      float v0 = SHm[m * 4 + 1], v1 = SHm[m * 4 + 2], v2 = SHm[m * 4 + 3];
      if (L == 0) {
        if (b == 0) acc_s[j] += A0 * ss * su * w;
        else { float f = A0 * su * w;
               acc_v1[j][0] += f * v0; acc_v1[j][1] += f * v1; acc_v1[j][2] += f * v2; }
      } else {
        float ox = VO[m * 48 + u * 3 + 0];
        float oy = VO[m * 48 + u * 3 + 1];
        float oz = VO[m * 48 + u * 3 + 2];
        if (b == 0) acc_s[j] += A0 * ss * su * w;
        else if (b == 1) acc_s[j] += (A0 / SQ3F) * (ox * v0 + oy * v1 + oz * v2) * w;
        else if (b == 2) { float f = A0 * su * w;
                           acc_v1[j][0] += f * v0; acc_v1[j][1] += f * v1; acc_v1[j][2] += f * v2; }
        else if (b == 3) { float f = A0 * ss * w;
                           acc_v1[j][0] += f * ox; acc_v1[j][1] += f * oy; acc_v1[j][2] += f * oz; }
        else { float f = (A2 / SQ2F) * w;
               acc_v2[j][0] += f * (oy * v2 - oz * v1);
               acc_v2[j][1] += f * (oz * v0 - ox * v2);
               acc_v2[j][2] += f * (ox * v1 - oy * v0); }
      }
    }
  }

  if (active) {
#pragma unroll
    for (int j = 0; j < 8; ++j) {
      int e = e0 + mbase + j;
      if (e >= E) continue;
      float* ap = acc + (size_t)dsti[e] * 112;
      atomicAdd(ap + ncol, acc_s[j]);
#pragma unroll
      for (int i = 0; i < 3; ++i) atomicAdd(ap + 16 + ncol * 3 + i, acc_v1[j][i]);
      if (L == 1) {
#pragma unroll
        for (int i = 0; i < 3; ++i) atomicAdd(ap + 64 + ncol * 3 + i, acc_v2[j][i]);
      }
    }
  }
}

// ---------------------------------------------------------------------------
// x[n] = pad(x[n]) + intra_sum/cnt + inter_sum/cnt
// ---------------------------------------------------------------------------
__global__ void k_update(float* __restrict__ x, const float* __restrict__ ai,
                         const float* __restrict__ ci, const float* __restrict__ ax,
                         const float* __restrict__ cx, int total) {
  int id = blockIdx.x * blockDim.x + threadIdx.x;
  if (id >= total) return;
  int n = id / 112;
  x[id] += ai[id] / fmaxf(ci[n], 1.0f) + ax[id] / fmaxf(cx[n], 1.0f);
}

// ---------------------------------------------------------------------------
// final conv (tpf): za(32) -> relu(@fconva 32x32) -> @fconvb 32x160, tensor
// product with Ws (E,16,2) blocks; scatter to agent accumulators.
// N-tile t: b = t/2, u = (t&1)*8 + (col/2), wcol = col&1.
// f16 LDS staging rows of 40 halves (80B, 16B-aligned), K=32 exactly.
// ---------------------------------------------------------------------------
__global__ __launch_bounds__(128) void k_fconv(
    const float* __restrict__ x, const float* __restrict__ zattr, const float* __restrict__ zsh,
    const int* __restrict__ zs, const int* __restrict__ zd,
    const _Float16* __restrict__ Wa, const float* __restrict__ ba,
    const _Float16* __restrict__ Wb, const float* __restrict__ bb,
    float* __restrict__ gacc, float* __restrict__ gcnt, int E) {
  // per-wave floats: EAh(320) Hh(320) S(256) SH(64) VO(768) VE(768)
  constexpr int WF = 320 + 320 + 256 + 64 + 768 + 768;
  __shared__ __align__(16) float lds[4 * WF];
  const int lane = threadIdx.x & 31;
  const int wv = threadIdx.x >> 5;
  _Float16* EAh = (_Float16*)(lds + wv * WF);          // [16][40] halves
  _Float16* Hh  = (_Float16*)(lds + wv * WF + 320);    // [16][40] halves
  float* S   = lds + wv * WF + 640;                    // [16][16]
  float* SHm = lds + wv * WF + 896;                    // [16][4]
  float* VO  = lds + wv * WF + 960;                    // [16][16][3]
  float* VE  = lds + wv * WF + 1728;                   // [16][16][3]
  int ntiles = (E + 15) >> 4;
  int tile = blockIdx.x * 4 + wv;
  bool active = (tile < ntiles);
  if (!active) tile = ntiles - 1;
  const int e0 = tile * 16;

  for (int idx = lane; idx < 512; idx += 32) {
    int m = idx >> 5, k = idx & 31;
    int e = e0 + m; if (e >= E) e = E - 1;
    float v;
    if (k < 16) v = zattr[(size_t)e * 16 + k];
    else { v = x[(size_t)zs[e] * 112 + (k - 16)]; S[m * 16 + (k - 16)] = v; }
    EAh[m * 40 + k] = (_Float16)v;
  }
  for (int idx = lane; idx < 768; idx += 32) {
    int m = idx / 48, k = idx - m * 48;
    int e = e0 + m; if (e >= E) e = E - 1;
    VO[idx] = x[(size_t)zs[e] * 112 + 16 + k];
    VE[idx] = x[(size_t)zs[e] * 112 + 64 + k];
  }
  for (int idx = lane; idx < 64; idx += 32) {
    int m = idx >> 2, c = idx & 3;
    int e = e0 + m; if (e >= E) e = E - 1;
    SHm[idx] = zsh[(size_t)e * 4 + c];
  }
  if (lane < 16) {
    int e = e0 + lane;
    if (active && e < E) atomicAdd(&gcnt[zd[e]], 1.0f);
  }
  __syncthreads();

  const int mrow = lane & 15;
  const int khalf = (lane >> 4) * 8;
  const int mbase = (lane < 16) ? 0 : 8;
  const int ncol = lane & 15;

  v16h a0 = frag_ld(EAh + mrow * 40 + khalf);
  for (int nt = 0; nt < 2; ++nt) {
    v8f c = {0.f, 0.f, 0.f, 0.f, 0.f, 0.f, 0.f, 0.f};
    const v16h b0 = *(const v16h*)(Wa + ((size_t)nt * 32 + lane) * 16);
    c = __builtin_amdgcn_wmma_f32_16x16x32_f16(false, a0, false, b0, (short)0, c, false, false);
    float bias = ba[nt * 16 + ncol];
#pragma unroll
    for (int j = 0; j < 8; ++j)
      Hh[(mbase + j) * 40 + nt * 16 + ncol] = (_Float16)fmaxf(c[j] + bias, 0.0f);
  }
  __syncthreads();
  v16h h0 = frag_ld(Hh + mrow * 40 + khalf);

  float acc_vo[8][3], acc_ve[8][3];
#pragma unroll
  for (int j = 0; j < 8; ++j)
#pragma unroll
    for (int i = 0; i < 3; ++i) { acc_vo[j][i] = 0.f; acc_ve[j][i] = 0.f; }
  const float AO = 0.14433756729740643f;  // 1/sqrt(48)
  const float AE = 0.17677669529663687f;  // 1/sqrt(32)

  for (int t = 0; t < 10; ++t) {
    v8f c = {0.f, 0.f, 0.f, 0.f, 0.f, 0.f, 0.f, 0.f};
    const v16h b0 = *(const v16h*)(Wb + ((size_t)t * 32 + lane) * 16);
    c = __builtin_amdgcn_wmma_f32_16x16x32_f16(false, h0, false, b0, (short)0, c, false, false);
    float bias = bb[t * 16 + ncol];
    int b = t >> 1;                       // uniform
    int u = (t & 1) * 8 + (ncol >> 1);    // per-lane
#pragma unroll
    for (int j = 0; j < 8; ++j) {
      int m = mbase + j;
      float w = c[j] + bias;
      float ss = SHm[m * 4 + 0];
      float v0 = SHm[m * 4 + 1], v1 = SHm[m * 4 + 2], v2 = SHm[m * 4 + 3];
      float ox = VO[m * 48 + u * 3 + 0], oy = VO[m * 48 + u * 3 + 1], oz = VO[m * 48 + u * 3 + 2];
      float ex = VE[m * 48 + u * 3 + 0], ey = VE[m * 48 + u * 3 + 1], ez = VE[m * 48 + u * 3 + 2];
      if (b == 0) { float f = AO * S[m * 16 + u] * w;
                    acc_vo[j][0] += f * v0; acc_vo[j][1] += f * v1; acc_vo[j][2] += f * v2; }
      else if (b == 1) { float f = AO * ss * w;
                         acc_vo[j][0] += f * ox; acc_vo[j][1] += f * oy; acc_vo[j][2] += f * oz; }
      else if (b == 2) { float f = (AO / SQ2F) * w;   // cr_e = cross(ve, vsh)/sqrt2
                         acc_vo[j][0] += f * (ey * v2 - ez * v1);
                         acc_vo[j][1] += f * (ez * v0 - ex * v2);
                         acc_vo[j][2] += f * (ex * v1 - ey * v0); }
      else if (b == 3) { float f = (AE / SQ2F) * w;   // cr_o = cross(vo, vsh)/sqrt2
                         acc_ve[j][0] += f * (oy * v2 - oz * v1);
                         acc_ve[j][1] += f * (oz * v0 - ox * v2);
                         acc_ve[j][2] += f * (ox * v1 - oy * v0); }
      else { float f = AE * ss * w;
             acc_ve[j][0] += f * ex; acc_ve[j][1] += f * ey; acc_ve[j][2] += f * ez; }
    }
  }

  if (active) {
    int wcol = ncol & 1;
#pragma unroll
    for (int j = 0; j < 8; ++j) {
      int e = e0 + mbase + j;
      if (e >= E) continue;
      float* gp = gacc + (size_t)zd[e] * 12;
#pragma unroll
      for (int i = 0; i < 3; ++i) {
        atomicAdd(gp + wcol * 3 + i, acc_vo[j][i]);
        atomicAdd(gp + 6 + wcol * 3 + i, acc_ve[j][i]);
      }
    }
  }
}

// ---------------------------------------------------------------------------
// head: g = gacc/cnt; tr/rot norms -> MLP(33->16->1) scale; write d_out
// ---------------------------------------------------------------------------
__global__ void k_head(const float* __restrict__ gacc, const float* __restrict__ gcnt,
                       const float* __restrict__ t_tr,
                       const float* __restrict__ tW1, const float* __restrict__ tb1,
                       const float* __restrict__ tW2, const float* __restrict__ tb2,
                       const float* __restrict__ rW1, const float* __restrict__ rb1,
                       const float* __restrict__ rW2, const float* __restrict__ rb2,
                       float* __restrict__ out, int A) {
  int a = blockIdx.x * blockDim.x + threadIdx.x;
  if (a >= A) return;
  float cn = fmaxf(gcnt[a], 1.0f);
  float g[12];
#pragma unroll
  for (int i = 0; i < 12; ++i) g[i] = gacc[a * 12 + i] / cn;
  float tr[3] = {g[0] + g[6], g[1] + g[7], g[2] + g[8]};
  float rt[3] = {g[3] + g[9], g[4] + g[10], g[5] + g[11]};
  float sg[32];
  const float c = -logf(10000.0f) / 16.0f;
  float tv = t_tr[a];
#pragma unroll
  for (int j = 0; j < 16; ++j) {
    float f = expf(c * (float)j);
    sg[j] = sinf(tv * f);
    sg[16 + j] = cosf(tv * f);
  }
  // tr branch
  float trn = sqrtf(tr[0] * tr[0] + tr[1] * tr[1] + tr[2] * tr[2]);
  float h[16];
#pragma unroll
  for (int j = 0; j < 16; ++j) h[j] = tb1[j] + trn * tW1[j];
  for (int k = 0; k < 32; ++k)
#pragma unroll
    for (int j = 0; j < 16; ++j) h[j] += sg[k] * tW1[(1 + k) * 16 + j];
  float s = tb2[0];
#pragma unroll
  for (int j = 0; j < 16; ++j) s += fmaxf(h[j], 0.0f) * tW2[j];
#pragma unroll
  for (int i = 0; i < 3; ++i) out[a * 3 + i] = tr[i] / trn * s;
  // rot branch
  float rn = sqrtf(rt[0] * rt[0] + rt[1] * rt[1] + rt[2] * rt[2]);
#pragma unroll
  for (int j = 0; j < 16; ++j) h[j] = rb1[j] + rn * rW1[j];
  for (int k = 0; k < 32; ++k)
#pragma unroll
    for (int j = 0; j < 16; ++j) h[j] += sg[k] * rW1[(1 + k) * 16 + j];
  float s2 = rb2[0];
#pragma unroll
  for (int j = 0; j < 16; ++j) s2 += fmaxf(h[j], 0.0f) * rW2[j];
#pragma unroll
  for (int i = 0; i < 3; ++i) out[A * 3 + a * 3 + i] = rt[i] / rn * s2;
}

// ---------------------------------------------------------------------------
extern "C" void kernel_launch(void* const* d_in, const int* in_sizes, int n_in,
                              void* d_out, int out_size, void* d_ws, size_t ws_size,
                              hipStream_t stream) {
  (void)n_in; (void)out_size; (void)ws_size;
  const float* x_feat = (const float*)d_in[0];
  const float* pos    = (const float*)d_in[1];
  const float* node_t = (const float*)d_in[2];
  const float* t_tr   = (const float*)d_in[3];
  const float* apos   = (const float*)d_in[4];
  const int* eidx = (const int*)d_in[5];
  const int* cidx = (const int*)d_in[6];
  const int* zidx = (const int*)d_in[7];
#define P(i) ((const float*)d_in[(i)])
  int N  = in_sizes[2];
  int A  = in_sizes[3];
  int E1 = in_sizes[5] / 2;
  int E2 = in_sizes[6] / 2;
  int NC = in_sizes[7] / 2;

  float* ws = (float*)d_ws;
  size_t off = 0;
  auto F = [&](size_t n) -> float* { float* p = ws + off; off += (n + 7) & ~(size_t)7; return p; };
  float* sig   = F((size_t)N * 32);
  float* x     = F((size_t)N * 112);
  float* acc_i = F((size_t)N * 112);
  float* cnt_i = F((size_t)N);
  float* acc_x = F((size_t)N * 112);
  float* cnt_x = F((size_t)N);
  float* eattr = F((size_t)E1 * 16);
  float* esh   = F((size_t)E1 * 4);
  float* cattr = F((size_t)E2 * 16);
  float* csh   = F((size_t)E2 * 4);
  float* zattr = F((size_t)NC * 16);
  float* zsh   = F((size_t)NC * 4);
  float* gacc  = F((size_t)A * 12);
  float* gcnt  = F((size_t)A);
  _Float16* hb = (_Float16*)(ws + off);
  size_t ho = 0;
  auto Hh = [&](size_t n) -> _Float16* { _Float16* p = hb + ho; ho += n; return p; };
  _Float16* pkA0  = Hh(3 * 2 * 512);
  _Float16* pkB0  = Hh(32 * 2 * 512);
  _Float16* pkXA0 = Hh(3 * 2 * 512);
  _Float16* pkXB0 = Hh(32 * 2 * 512);
  _Float16* pkA1  = Hh(3 * 2 * 512);
  _Float16* pkB1  = Hh(80 * 2 * 512);
  _Float16* pkXA1 = Hh(3 * 2 * 512);
  _Float16* pkXB1 = Hh(80 * 2 * 512);
  _Float16* pkFA  = Hh(2 * 1 * 512);
  _Float16* pkFB  = Hh(10 * 1 * 512);

  hipMemsetAsync(acc_i, 0, (size_t)N * 112 * 4, stream);
  hipMemsetAsync(cnt_i, 0, (size_t)N * 4, stream);
  hipMemsetAsync(acc_x, 0, (size_t)N * 112 * 4, stream);
  hipMemsetAsync(cnt_x, 0, (size_t)N * 4, stream);
  hipMemsetAsync(gacc, 0, (size_t)A * 12 * 4, stream);
  hipMemsetAsync(gcnt, 0, (size_t)A * 4, stream);

  auto pack = [&](const float* W, int K, int ncols, _Float16* o) {
    int ks = (K + 31) / 32;
    int tot = (ncols / 16) * ks * 512;
    k_pack<<<(tot + 255) / 256, 256, 0, stream>>>(W, K, ncols, ks, o, tot);
  };
  pack(P(24), 48, 48, pkA0);   pack(P(26), 48, 512, pkB0);
  pack(P(28), 48, 48, pkXA0);  pack(P(30), 48, 512, pkXB0);
  pack(P(32), 48, 48, pkA1);   pack(P(34), 48, 1280, pkB1);
  pack(P(36), 48, 48, pkXA1);  pack(P(38), 48, 1280, pkXB1);
  pack(P(40), 32, 32, pkFA);   pack(P(42), 32, 160, pkFB);

  k_sig<<<(N + 127) / 128, 128, 0, stream>>>(node_t, sig, N);
  k_node_mlp<<<(N + 127) / 128, 128, 0, stream>>>(x_feat, sig, P(8), P(9), P(10), P(11), x, N);

  k_edge_attr<<<(E1 + 127) / 128, 128, 0, stream>>>(pos, pos, sig, eidx, eidx + E1,
      P(12), P(13), P(14), P(15), eattr, esh, E1, 1, 0);
  k_edge_attr<<<(E2 + 127) / 128, 128, 0, stream>>>(pos, pos, sig, cidx, cidx + E2,
      P(16), P(17), P(18), P(19), cattr, csh, E2, 0, 0);
  k_edge_attr<<<(NC + 127) / 128, 128, 0, stream>>>(pos, apos, sig, zidx, zidx + NC,
      P(20), P(21), P(22), P(23), zattr, zsh, NC, 0, 1);

  int t1 = (E1 + 15) / 16, t2 = (E2 + 15) / 16, tz = (NC + 15) / 16;

  // layer 0
  k_conv<0><<<(t1 + 3) / 4, 128, 0, stream>>>(x, eattr, esh, eidx, eidx + E1,
      pkA0, P(25), pkB0, P(27), acc_i, cnt_i, E1);
  k_conv<0><<<(t2 + 3) / 4, 128, 0, stream>>>(x, cattr, csh, cidx, cidx + E2,
      pkXA0, P(29), pkXB0, P(31), acc_x, cnt_x, E2);
  k_update<<<((N * 112) + 255) / 256, 256, 0, stream>>>(x, acc_i, cnt_i, acc_x, cnt_x, N * 112);

  hipMemsetAsync(acc_i, 0, (size_t)N * 112 * 4, stream);
  hipMemsetAsync(cnt_i, 0, (size_t)N * 4, stream);
  hipMemsetAsync(acc_x, 0, (size_t)N * 112 * 4, stream);
  hipMemsetAsync(cnt_x, 0, (size_t)N * 4, stream);

  // layer 1
  k_conv<1><<<(t1 + 3) / 4, 128, 0, stream>>>(x, eattr, esh, eidx, eidx + E1,
      pkA1, P(33), pkB1, P(35), acc_i, cnt_i, E1);
  k_conv<1><<<(t2 + 3) / 4, 128, 0, stream>>>(x, cattr, csh, cidx, cidx + E2,
      pkXA1, P(37), pkXB1, P(39), acc_x, cnt_x, E2);
  k_update<<<((N * 112) + 255) / 256, 256, 0, stream>>>(x, acc_i, cnt_i, acc_x, cnt_x, N * 112);

  // final conv + heads
  k_fconv<<<(tz + 3) / 4, 128, 0, stream>>>(x, zattr, zsh, zidx, zidx + NC,
      pkFA, P(41), pkFB, P(43), gacc, gcnt, NC);
  k_head<<<(A + 63) / 64, 64, 0, stream>>>(gacc, gcnt, t_tr,
      P(44), P(45), P(46), P(47), P(48), P(49), P(50), P(51), (float*)d_out, A);
#undef P
}
